// EnhancedEdgeScorer_38293928411614
// MI455X (gfx1250) — matrix operand order repaired and checked by reading, not compiled, not run on hardware
//
#include <hip/hip_runtime.h>
#include <hip/hip_bf16.h>
#include <math.h>

// ---------------------------------------------------------------------------
// EnhancedEdgeScorer for MI455X (gfx1250, wave32, WMMA f32_16x16x32_f16)
// f16 column-major weights: B fragment = one contiguous 32B chunk.
// f16 activation mirrors (xh/oh): A-tile staging = pure 16B copies, done with
// GLOBAL_LOAD_ASYNC_TO_LDS_B128 (ASYNCcnt-tracked DMA).
// ---------------------------------------------------------------------------

typedef __attribute__((ext_vector_type(16))) _Float16 v16h;
typedef __attribute__((ext_vector_type(8)))  _Float16 v8h;
typedef __attribute__((ext_vector_type(8)))  float    v8f;
typedef int v4i __attribute__((vector_size(16)));

#define HID 128
#define KNBR 16
#define CTX 17   // self + 16 neighbors

#if defined(__has_builtin)
#  if __has_builtin(__builtin_amdgcn_global_load_async_to_lds_b128)
#    define HAVE_ASYNC_LDS 1
#  endif
#  if __has_builtin(__builtin_amdgcn_s_wait_asynccnt)
#    define HAVE_WAIT_ASYNC 1
#  endif
#endif

typedef __attribute__((address_space(1))) v4i glb_v4i;
typedef __attribute__((address_space(3))) v4i lds_v4i;

// 16-byte global->LDS copy (async DMA when available)
static __device__ __forceinline__ void copy16_to_lds(const _Float16* g,
                                                     _Float16* l)
{
#ifdef HAVE_ASYNC_LDS
    __builtin_amdgcn_global_load_async_to_lds_b128(
        (glb_v4i*)g, (lds_v4i*)l, /*offset=*/0, /*cpol=*/0);
#else
    *(v8h*)l = *(const v8h*)g;
#endif
}

static __device__ __forceinline__ void async_wait()
{
#ifdef HAVE_ASYNC_LDS
#  ifdef HAVE_WAIT_ASYNC
    __builtin_amdgcn_s_wait_asynccnt(0);
#  else
    asm volatile("s_wait_asynccnt 0x0" ::: "memory");
#  endif
#endif
}

static __device__ __forceinline__ v8f wmma_f16(v16h a, v16h b, v8f c) {
    return __builtin_amdgcn_wmma_f32_16x16x32_f16(
        /*neg_a=*/false, a, /*neg_b=*/false, b,
        /*c_mod=*/(short)0, c, /*reuse_a=*/false, /*reuse_b=*/false);
}

// A fragment: 16 contiguous f16 from an f16 buffer (LDS or global)
static __device__ __forceinline__ v16h load_afrag(const _Float16* p) {
    return *(const v16h*)(p);
}

// B fragment: 16 contiguous f16 (column-major weight, fixed col, k contiguous)
static __device__ __forceinline__ v16h load_bfrag(const _Float16* p) {
    return *(const v16h*)(p);
}

// ---------------------------------------------------------------------------
// Weight transpose + f32->f16: dst[col*K + k] = src[k*Ncol + col]
// ---------------------------------------------------------------------------
__global__ void k_w2h(const float* __restrict__ src, _Float16* __restrict__ dst,
                      int K, int Ncol)
{
    const int idx = blockIdx.x * blockDim.x + threadIdx.x;
    if (idx >= K * Ncol) return;
    const int k = idx / Ncol, col = idx % Ncol;
    dst[(size_t)col * K + k] = (_Float16)src[idx];
}

// ---------------------------------------------------------------------------
// Node encoder: x/xh[M x 128] = feats[M x 160] @ proj_W + proj_b
// feats = concat(type_embed, cat0, cat1, relu(deg)); gathered once into LDS.
// ---------------------------------------------------------------------------
__global__ __launch_bounds__(128)
void k_encode(const int* __restrict__ type_idx, const int* __restrict__ cat_idx,
              const float* __restrict__ log_deg,
              const float* __restrict__ te, const float* __restrict__ c0,
              const float* __restrict__ c1,
              const float* __restrict__ dW, const float* __restrict__ db,
              const _Float16* __restrict__ projWt,
              const float* __restrict__ projb,
              float* __restrict__ x, _Float16* __restrict__ xh, int M)
{
    __shared__ _Float16 sA[16 * 160];    // 5 KB

    const int t  = threadIdx.x;          // 0..127
    const int m0 = blockIdx.x * 16;

    // stage gathered features for 16 nodes
    for (int idx = t; idx < 16 * 160; idx += 128) {
        const int r = idx / 160, k = idx % 160;
        int node = m0 + r; if (node >= M) node = M - 1;
        float v;
        if (k < 64)       v = te[(size_t)type_idx[node] * 64 + k];
        else if (k < 96)  v = c0[(size_t)cat_idx[node * 2 + 0] * 32 + (k - 64)];
        else if (k < 128) v = c1[(size_t)cat_idx[node * 2 + 1] * 32 + (k - 96)];
        else {
            v = log_deg[node] * dW[k - 128] + db[k - 128];
            v = fmaxf(v, 0.f);
        }
        sA[idx] = (_Float16)v;
    }
    __syncthreads();

    const int lane = t & 31;
    const int wave = t >> 5;
    const int row  = lane & 15;
    const int hi   = (lane >> 4) & 1;
    const int kbA  = hi ? 16 : 0;
    const int kbB  = hi ? 16 : 0;
    const int mb   = hi ? 8 : 0;

#pragma unroll
    for (int w = 0; w < 2; ++w) {
        const int bcol = (wave * 2 + w) * 16 + row;
        const _Float16* bp = projWt + (size_t)bcol * 160;
        __builtin_prefetch(bp, 0, 3);
        v8f acc = {};
#pragma unroll
        for (int kt = 0; kt < 160; kt += 32) {
            v16h a = load_afrag(sA + row * 160 + kt + kbA);
            v16h b = load_bfrag(bp + kt + kbB);
            acc = wmma_f16(a, b, acc);
        }
        const float bs = projb[bcol];
        float*    op = x  + (size_t)(m0 + mb) * HID + bcol;
        _Float16* hp = xh + (size_t)(m0 + mb) * HID + bcol;
        if (m0 + 16 <= M) {
#pragma unroll
            for (int r8 = 0; r8 < 8; ++r8) {
                const float v = acc[r8] + bs;
                op[(size_t)r8 * HID] = v;
                hp[(size_t)r8 * HID] = (_Float16)v;
            }
        } else {
#pragma unroll
            for (int r8 = 0; r8 < 8; ++r8) {
                if (m0 + mb + r8 < M) {
                    const float v = acc[r8] + bs;
                    op[(size_t)r8 * HID] = v;
                    hp[(size_t)r8 * HID] = (_Float16)v;
                }
            }
        }
    }
}

// ---------------------------------------------------------------------------
// GEMM (K=128): out[16 x 64-slice] = ep(Ah @ W + bias)
// 128 threads = 4 waves, one 16-wide n-tile per wave; f16 A tile async-staged
// in LDS; B from f16 column-major weights (col stride 128).
// EP: 0 = store fp32, 2 = x/xh = relu(has_nbr ? v : x_old)
// ---------------------------------------------------------------------------
template <int EP, int LDO>
__global__ __launch_bounds__(128)
void k_gemm_t(const _Float16* __restrict__ Ah, const _Float16* __restrict__ Wt,
              const float* __restrict__ bias, float* __restrict__ out,
              int M,
              const float* __restrict__ x_old, _Float16* __restrict__ xh,
              const int* __restrict__ hnb)
{
    __shared__ _Float16 sA[16 * HID];    // 4 KB

    const int t     = threadIdx.x;       // 0..127
    const int m0    = blockIdx.x * 16;
    const int cbase = blockIdx.y * 64;

    // stage A tile (16 rows x 128 f16 = 256 chunks of 16B; 2 per thread)
#pragma unroll
    for (int j = 0; j < 2; ++j) {
        const int chunk = t + j * 128;   // 0..255
        const int r = chunk >> 4;
        const int c = (chunk & 15) * 8;
        int ar = m0 + r; if (ar >= M) ar = M - 1;
        copy16_to_lds(Ah + (size_t)ar * HID + c, sA + r * HID + c);
    }
    async_wait();
    __syncthreads();

    const int lane = t & 31;
    const int wave = t >> 5;
    const int row  = lane & 15;
    const int hi   = (lane >> 4) & 1;
    const int kbA  = hi ? 16 : 0;
    const int kbB  = hi ? 16 : 0;
    const int mb   = hi ? 8 : 0;
    const int gcol = cbase + wave * 16 + row;
    const _Float16* bp = Wt + (size_t)gcol * HID;
    __builtin_prefetch(bp, 0, 3);

    v8f acc = {};
#pragma unroll
    for (int kt = 0; kt < HID; kt += 32) {
        v16h a = load_afrag(sA + row * HID + kt + kbA);
        v16h b = load_bfrag(bp + kt + kbB);
        acc = wmma_f16(a, b, acc);
    }

    const float bs = bias[gcol];
    float* op = out + (size_t)(m0 + mb) * LDO + gcol;
    if (m0 + 16 <= M) {
#pragma unroll
        for (int r8 = 0; r8 < 8; ++r8) {
            float v = acc[r8] + bs;
            if (EP == 2) {
                const float xo = x_old[(size_t)(m0 + mb + r8) * LDO + gcol];
                v = hnb[m0 + mb + r8] ? v : xo;
                v = fmaxf(v, 0.f);
                xh[(size_t)(m0 + mb + r8) * LDO + gcol] = (_Float16)v;
            }
            op[(size_t)r8 * LDO] = v;
        }
    } else {
#pragma unroll
        for (int r8 = 0; r8 < 8; ++r8) {
            const int m = m0 + mb + r8;
            if (m >= M) continue;
            float v = acc[r8] + bs;
            if (EP == 2) {
                const float xo = x_old[(size_t)m * LDO + gcol];
                v = hnb[m] ? v : xo;
                v = fmaxf(v, 0.f);
                xh[(size_t)m * LDO + gcol] = (_Float16)v;
            }
            op[(size_t)r8 * LDO] = v;
        }
    }
}

// ---------------------------------------------------------------------------
// has_nbr precompute
// ---------------------------------------------------------------------------
__global__ void k_hasnbr(const int* __restrict__ nbr_mask,
                         int* __restrict__ hnb, int N)
{
    int n = blockIdx.x * blockDim.x + threadIdx.x;
    if (n >= N) return;
    int any = 0;
#pragma unroll
    for (int k = 0; k < KNBR; ++k) any |= nbr_mask[(size_t)n * KNBR + k];
    hnb[n] = any ? 1 : 0;
}

// ---------------------------------------------------------------------------
// Single-query MHA over local context; 4 nodes per block (wave per node),
// lane == head dim (D=32 == wave32). Output written as f16 (GEMM input).
// ---------------------------------------------------------------------------
__global__ __launch_bounds__(128)
void k_attn(const float* __restrict__ qkv,
            const int* __restrict__ nbr_idx, const int* __restrict__ nbr_mask,
            _Float16* __restrict__ oh, int N)
{
    __shared__ float sc[4][4][CTX];
    __shared__ int cidx[4][CTX];

    const int wave = threadIdx.x >> 5;
    const int lane = threadIdx.x & 31;
    int n = blockIdx.x * 4 + wave; if (n >= N) n = N - 1;
    const float scale = 0.1767766952966369f;   // 1/sqrt(32)

    if (lane < CTX)
        cidx[wave][lane] = (lane == 0) ? n
                                       : nbr_idx[(size_t)n * KNBR + (lane - 1)];
    __syncthreads();

    const size_t qb = (size_t)n * 384;
    float q0 = qkv[qb +   0 + lane];
    float q1 = qkv[qb +  32 + lane];
    float q2 = qkv[qb +  64 + lane];
    float q3 = qkv[qb +  96 + lane];

    for (int c = 0; c < CTX; ++c) {
        const size_t kb = (size_t)cidx[wave][c] * 384 + 128;
        float p0 = q0 * qkv[kb +   0 + lane];
        float p1 = q1 * qkv[kb +  32 + lane];
        float p2 = q2 * qkv[kb +  64 + lane];
        float p3 = q3 * qkv[kb +  96 + lane];
#pragma unroll
        for (int off = 16; off > 0; off >>= 1) {
            p0 += __shfl_xor(p0, off);
            p1 += __shfl_xor(p1, off);
            p2 += __shfl_xor(p2, off);
            p3 += __shfl_xor(p3, off);
        }
        if (lane == 0) {
            const bool msk = (c == 0) ||
                             (nbr_mask[(size_t)n * KNBR + (c - 1)] != 0);
            sc[wave][0][c] = msk ? p0 * scale : -1e9f;
            sc[wave][1][c] = msk ? p1 * scale : -1e9f;
            sc[wave][2][c] = msk ? p2 * scale : -1e9f;
            sc[wave][3][c] = msk ? p3 * scale : -1e9f;
        }
    }
    __syncthreads();

#pragma unroll
    for (int h = 0; h < 4; ++h) {
        float mx = -1e30f;
        for (int c = 0; c < CTX; ++c) mx = fmaxf(mx, sc[wave][h][c]);
        float sum = 0.f;
        for (int c = 0; c < CTX; ++c) sum += __expf(sc[wave][h][c] - mx);
        const float inv = 1.f / sum;
        float accv = 0.f;
        for (int c = 0; c < CTX; ++c) {
            const float w = __expf(sc[wave][h][c] - mx) * inv;
            accv += w * qkv[(size_t)cidx[wave][c] * 384 + 256 + h * 32 + lane];
        }
        oh[(size_t)n * HID + h * 32 + lane] = (_Float16)accv;
    }
}

// ---------------------------------------------------------------------------
// Fused edge-scoring MLP per 16-edge tile; 128 threads = 4 waves.
// Gathered [xh_src | xh_dst] async-staged to LDS (pure 16B copies).
// ---------------------------------------------------------------------------
__global__ __launch_bounds__(128)
void k_edge(const _Float16* __restrict__ xh, const int* __restrict__ edges,
            const float* __restrict__ ef,
            const _Float16* __restrict__ e1Wt, const float* __restrict__ eW1,
            const float* __restrict__ eb1,
            const _Float16* __restrict__ e2Wt, const float* __restrict__ eb2,
            const float* __restrict__ eW3, const float* __restrict__ eb3,
            float* __restrict__ out, int E)
{
    __shared__ _Float16 sIn[16 * 256];   // 8 KB
    __shared__ _Float16 sH1[16 * HID];   // 4 KB
    __shared__ float    sH2[16][64];     // 4 KB
    __shared__ int      esrc[16];
    __shared__ int      edst[16];

    const int t  = threadIdx.x;          // 0..127
    const int e0 = blockIdx.x * 16;

    if (t < 16) {
        esrc[t] = edges[(size_t)(e0 + t) * 2 + 0];
        edst[t] = edges[(size_t)(e0 + t) * 2 + 1];
    }
    __syncthreads();

    // stage [xh_src | xh_dst]: 16 edges x 512B = 512 chunks of 16B; 4/thread
#pragma unroll
    for (int j = 0; j < 4; ++j) {
        const int chunk = t + j * 128;       // 0..511
        const int r  = chunk >> 5;           // edge in tile
        const int cc = chunk & 31;           // 16B chunk within 512B row
        const int node = (cc < 16) ? esrc[r] : edst[r];
        const int c = (cc & 15) * 8;         // f16 col within node row
        copy16_to_lds(xh + (size_t)node * HID + c, sIn + r * 256 + cc * 8);
    }
    async_wait();
    __syncthreads();

    const int lane = t & 31;
    const int wave = t >> 5;
    const int row  = lane & 15;
    const int hi   = (lane >> 4) & 1;
    const int kbA  = hi ? 16 : 0;
    const int kbB  = hi ? 16 : 0;
    const int mb   = hi ? 8 : 0;

    // ---- layer 1: 16x258 @ 258x128; wave handles n-tiles wave*2, wave*2+1
#pragma unroll
    for (int w = 0; w < 2; ++w) {
        const int bcol = (wave * 2 + w) * 16 + row;
        const _Float16* bp = e1Wt + (size_t)bcol * 256;
        __builtin_prefetch(bp, 0, 3);
        v8f acc = {};
#pragma unroll
        for (int kt = 0; kt < 256; kt += 32) {
            v16h a = load_afrag(sIn + row * 256 + kt + kbA);
            v16h b = load_bfrag(bp + kt + kbB);
            acc = wmma_f16(a, b, acc);
        }
#pragma unroll
        for (int r8 = 0; r8 < 8; ++r8) {
            const int m = mb + r8;
            float v = acc[r8]
                + ef[(size_t)(e0 + m) * 2 + 0] * eW1[(size_t)256 * HID + bcol]
                + ef[(size_t)(e0 + m) * 2 + 1] * eW1[(size_t)257 * HID + bcol]
                + eb1[bcol];
            sH1[m * HID + bcol] = (_Float16)fmaxf(v, 0.f);
        }
    }
    __syncthreads();

    // ---- layer 2: 16x128 @ 128x64; wave handles n-tile `wave`
    {
        const int bcol = wave * 16 + row;
        const _Float16* bp = e2Wt + (size_t)bcol * HID;
        v8f acc = {};
#pragma unroll
        for (int kt = 0; kt < HID; kt += 32) {
            v16h a = load_afrag(sH1 + row * HID + kt + kbA);
            v16h b = load_bfrag(bp + kt + kbB);
            acc = wmma_f16(a, b, acc);
        }
#pragma unroll
        for (int r8 = 0; r8 < 8; ++r8)
            sH2[mb + r8][bcol] = fmaxf(acc[r8] + eb2[bcol], 0.f);
    }
    __syncthreads();

    // ---- layer 3: 16x64 @ 64x1
    if (t < 16) {
        float s = eb3[0];
#pragma unroll
        for (int j = 0; j < 64; ++j) s += sH2[t][j] * eW3[j];
        if (e0 + t < E) out[e0 + t] = s;
    }
}

// ---------------------------------------------------------------------------
// Host launcher
// ---------------------------------------------------------------------------
extern "C" void kernel_launch(void* const* d_in, const int* in_sizes, int n_in,
                              void* d_out, int out_size, void* d_ws, size_t ws_size,
                              hipStream_t stream)
{
    const int*   type_idx  = (const int*)d_in[0];
    const int*   cat_idx   = (const int*)d_in[1];
    const int*   nbr_idx   = (const int*)d_in[2];
    const int*   nbr_mask  = (const int*)d_in[3];
    const int*   edges     = (const int*)d_in[4];
    const float* log_deg   = (const float*)d_in[5];
    const float* edge_feats= (const float*)d_in[6];
    const float* type_embed= (const float*)d_in[7];
    const float* cat0      = (const float*)d_in[8];
    const float* cat1      = (const float*)d_in[9];
    const float* deg_W     = (const float*)d_in[10];
    const float* deg_b     = (const float*)d_in[11];
    const float* proj_W    = (const float*)d_in[12];
    const float* proj_b    = (const float*)d_in[13];
    const float* Wqkv      = (const float*)d_in[14];
    const float* bqkv      = (const float*)d_in[15];
    const float* Wo        = (const float*)d_in[16];
    const float* bo        = (const float*)d_in[17];
    const float* eW1       = (const float*)d_in[18];
    const float* eb1       = (const float*)d_in[19];
    const float* eW2       = (const float*)d_in[20];
    const float* eb2       = (const float*)d_in[21];
    const float* eW3       = (const float*)d_in[22];
    const float* eb3       = (const float*)d_in[23];

    const int N = in_sizes[0];
    const int E = in_sizes[4] / 2;

    float*    x   = (float*)d_ws;                    // N x 128 fp32
    float*    qkv = x + (size_t)N * HID;             // N x 384 fp32
    _Float16* xh  = (_Float16*)(qkv + (size_t)N * 384);  // N x 128 f16
    _Float16* oh  = xh + (size_t)N * HID;            // N x 128 f16
    int*      hnb = (int*)(oh + (size_t)N * HID);    // N

    // f16 column-major weight buffers (256B-aligned region after hnb)
    uintptr_t wp = (uintptr_t)(hnb + N);
    wp = (wp + 255) & ~(uintptr_t)255;
    _Float16* projWt = (_Float16*)wp;                // 128 cols x 160
    _Float16* qkvWt  = projWt + 160 * 128;           // 3 x (384 cols x 128)
    _Float16* woWt   = qkvWt + (size_t)3 * 384 * HID;
    _Float16* e1Wt   = woWt + (size_t)3 * HID * HID; // 128 cols x 256
    _Float16* e2Wt   = e1Wt + 256 * 128;             // 64 cols x 128

    // --- one-time weight transpose/convert (tiny) ---
    k_w2h<<<(160 * 128 + 255) / 256, 256, 0, stream>>>(proj_W, projWt, 160, 128);
    for (int l = 0; l < 3; ++l) {
        k_w2h<<<(HID * 384 + 255) / 256, 256, 0, stream>>>(
            Wqkv + (size_t)l * HID * 384, qkvWt + (size_t)l * 384 * HID, HID, 384);
        k_w2h<<<(HID * HID + 255) / 256, 256, 0, stream>>>(
            Wo + (size_t)l * HID * HID, woWt + (size_t)l * HID * HID, HID, HID);
    }
    k_w2h<<<(256 * 128 + 255) / 256, 256, 0, stream>>>(eW1, e1Wt, 256, 128);
    k_w2h<<<(HID * 64 + 255) / 256, 256, 0, stream>>>(eW2, e2Wt, HID, 64);

    k_hasnbr<<<(N + 255) / 256, 256, 0, stream>>>(nbr_mask, hnb, N);

    const int mt = (N + 15) / 16;

    // node encoder: x/xh = feats(160) @ proj_W + proj_b
    k_encode<<<mt, 128, 0, stream>>>(
        type_idx, cat_idx, log_deg, type_embed, cat0, cat1, deg_W, deg_b,
        projWt, proj_b, x, xh, N);

    for (int l = 0; l < 3; ++l) {
        // qkv = xh @ Wqkv[l] + bqkv[l]   (Nout=384 -> grid.y = 6)
        k_gemm_t<0, 384><<<dim3(mt, 6), 128, 0, stream>>>(
            xh, qkvWt + (size_t)l * 384 * HID, bqkv + (size_t)l * 384, qkv,
            N, nullptr, nullptr, nullptr);

        // local attention -> oh (f16)
        k_attn<<<(N + 3) / 4, 128, 0, stream>>>(qkv, nbr_idx, nbr_mask, oh, N);

        // x/xh = relu(where(has_nbr, oh @ Wo[l] + bo[l], x))
        k_gemm_t<2, HID><<<dim3(mt, 2), 128, 0, stream>>>(
            oh, woWt + (size_t)l * HID * HID, bo + (size_t)l * HID, x,
            N, x, xh, hnb);
    }

    // fused edge MLP -> logits
    k_edge<<<(E + 15) / 16, 128, 0, stream>>>(
        xh, edges, edge_feats, e1Wt, eW1, eb1, e2Wt, eb2, eW3, eb3,
        (float*)d_out, E);
}
